// VGG16_LSTM_44083544326799
// MI455X (gfx1250) — matrix operand order repaired and checked by reading, not compiled, not run on hardware
//
#include <hip/hip_runtime.h>

typedef __attribute__((ext_vector_type(16))) __bf16 v16bf;
typedef __attribute__((ext_vector_type(8)))  __bf16 v8bf;
typedef __attribute__((ext_vector_type(4)))  __bf16 v4bf;
typedef __attribute__((ext_vector_type(8)))  float  v8f;

// Problem sizes
#define BB   256
#define TT   25
#define DD   4096
#define HH   512
#define GG   2048          // 4*H
#define MR   (BB*TT)       // 6400 rows of the big GEMM

#if defined(__HIP_DEVICE_COMPILE__) && __has_builtin(__builtin_amdgcn_tensor_load_to_lds)
#define USE_TDM 1
#else
#define USE_TDM 0
#endif

#if USE_TDM
typedef __attribute__((ext_vector_type(4))) unsigned int v4u_t;
typedef __attribute__((ext_vector_type(8))) int v8i_t;
typedef __attribute__((ext_vector_type(4))) int v4i_t;

// Low 32 bits of a flat pointer to LDS == byte offset within LDS
// (the LDS aperture lives entirely in addr[63:32]).
__device__ __forceinline__ unsigned lds_offset(const void* p) {
  return (unsigned)(size_t)p;
}

// Issue one 2-D TDM tile load: tile_rows rows of tile_k bf16 elements,
// row stride row_stride_elems in global memory.  LDS destination gets a
// 16-byte pad after every 64 bytes (pad_interval=16 DW, pad_amount=4 DW),
// matching the conflict-free 80-byte LDS row stride used by the WMMA
// fragment loaders.  Tracked with TENSORcnt.
__device__ __forceinline__ void tdm_load_tile_2d(unsigned lds_addr, const void* gptr,
                                                 unsigned tile_k, unsigned tile_rows,
                                                 unsigned long long row_stride_elems) {
  const unsigned long long ga = (unsigned long long)(size_t)gptr;
  v4u_t g0;
  g0[0] = 1u;                                                   // count=1 (user D#)
  g0[1] = lds_addr;                                             // lds_addr[31:0]
  g0[2] = (unsigned)(ga & 0xFFFFFFFFull);                       // global_addr lo
  g0[3] = (unsigned)((ga >> 32) & 0x01FFFFFFull) | 0x80000000u; // addr hi | type=2
  v8i_t g1;
  g1[0] = (int)((1u << 16)      // data_size = 2 bytes
              | (1u << 20)      // pad_enable
              | (3u << 22)      // pad_interval: 16 DWORDs (64B)
              | (3u << 25));    // pad_amount:   4 DWORDs (16B)
  g1[1] = (int)(tile_k << 16);                                  // tensor_dim0[15:0]
  g1[2] = (int)(tile_rows << 16);                               // dim0 hi | tensor_dim1 lo
  g1[3] = (int)(tile_k << 16);                                  // dim1 hi | tile_dim0
  g1[4] = (int)tile_rows;                                       // tile_dim1 | tile_dim2=0
  g1[5] = (int)(unsigned)(row_stride_elems & 0xFFFFFFFFull);    // dim0_stride lo32
  g1[6] = (int)(unsigned)((row_stride_elems >> 32) & 0xFFFFull);// dim0_stride hi16
  g1[7] = 0;
  const v4i_t z4 = {0, 0, 0, 0};
#if __has_include(<hip/amd_detail/amd_gfx1250_TDM.h>)
  const v8i_t z8 = {0, 0, 0, 0, 0, 0, 0, 0};
  __builtin_amdgcn_tensor_load_to_lds(g0, g1, z4, z4, z8, 0);
#else
  __builtin_amdgcn_tensor_load_to_lds(g0, g1, z4, z4, 0);
#endif
}
#endif // USE_TDM

// ---------------------------------------------------------------------------
// 1) Strip channel 0 of x and convert fp32 -> bf16.  out row = b*25 + t.
// ---------------------------------------------------------------------------
__global__ __launch_bounds__(256) void k_convert_x(const float* __restrict__ x,
                                                   __bf16* __restrict__ xb) {
  size_t g = (size_t)blockIdx.x * 256 + threadIdx.x;   // one float4 per thread
  size_t e = g * 4;
  size_t row = e >> 12;          // / 4096
  size_t d   = e & 4095;
  size_t b   = row / 25u;
  size_t t   = row - b * 25u;
  const float4 v = *(const float4*)(x + ((b * 50u + t) << 12) + d); // (b,0,t,:)
  v4bf o;
  o[0] = (__bf16)v.x; o[1] = (__bf16)v.y; o[2] = (__bf16)v.z; o[3] = (__bf16)v.w;
  *(v4bf*)(xb + e) = o;
}

// ---------------------------------------------------------------------------
// 2) Transpose + convert weights: out[c*R + r] = (bf16) in[r*C + c]
// ---------------------------------------------------------------------------
__global__ __launch_bounds__(256) void k_transpose_bf16(const float* __restrict__ in,
                                                        __bf16* __restrict__ out,
                                                        int R, int C) {
  __shared__ float tile[32][33];
  const int tx = threadIdx.x & 31;
  const int ty = threadIdx.x >> 5;          // 0..7
  const int c0 = blockIdx.x * 32;
  const int r0 = blockIdx.y * 32;
#pragma unroll
  for (int i = 0; i < 32; i += 8)
    tile[ty + i][tx] = in[(size_t)(r0 + ty + i) * C + (c0 + tx)];
  __syncthreads();
#pragma unroll
  for (int i = 0; i < 32; i += 8)
    out[(size_t)(c0 + ty + i) * R + (r0 + tx)] = (__bf16)tile[tx][ty + i];
}

// ---------------------------------------------------------------------------
// WMMA fragment helpers.  LDS tiles are [rows][40] bf16 (80B stride: 16B
// aligned, conflict-free for b128 reads).
//   A frag (16x32): lane l, elem j -> k = j + 16*(l>=16)
//   B frag (32x16): lane l, elem j -> k = j + 16*(l>=16)
// ---------------------------------------------------------------------------
__device__ __forceinline__ v16bf load_afrag(const __bf16* base, int kb) {
  v8bf lo = *(const v8bf*)(base + kb);
  v8bf hi = *(const v8bf*)(base + kb + 8);
  return __builtin_shufflevector(lo, hi, 0,1,2,3,4,5,6,7,8,9,10,11,12,13,14,15);
}
__device__ __forceinline__ v16bf load_bfrag(const __bf16* base, int kbB) {
  v8bf lo = *(const v8bf*)(base + kbB);
  v8bf hi = *(const v8bf*)(base + kbB + 8);
  return __builtin_shufflevector(lo, hi, 0,1,2,3,4,5,6,7,8,9,10,11,12,13,14,15);
}

// ---------------------------------------------------------------------------
// 3) Big GEMM: Z[m][n] = sum_k A[m][k]*BT[n][k] + bias[n]
//    M=6400, N=2048, K=4096.  WG tile 128x128, 8 waves (4x2), wave tile 32x64.
//    Tiles streamed into LDS by the Tensor Data Mover, double-buffered with a
//    single barrier per k-step (TDM of tile i+1 overlaps WMMAs of tile i).
// ---------------------------------------------------------------------------
__global__ __launch_bounds__(256) void k_gemm_zx(const __bf16* __restrict__ A,
                                                 const __bf16* __restrict__ BT,
                                                 const float* __restrict__ bias,
                                                 float* __restrict__ Z) {
#if USE_TDM
  __shared__ __align__(16) __bf16 As[2][128 * 40];
  __shared__ __align__(16) __bf16 Bs[2][128 * 40];
#else
  __shared__ __align__(16) __bf16 As[1][128 * 40];
  __shared__ __align__(16) __bf16 Bs[1][128 * 40];
#endif

  const int tid  = threadIdx.x;
  const int lane = tid & 31;
  const int wave = tid >> 5;
  const int wm   = wave & 3;          // 0..3 (rows)
  const int wn   = wave >> 2;         // 0..1 (cols)
  const int gm0  = blockIdx.y * 128;
  const int gn0  = blockIdx.x * 128;
  const int l15  = lane & 15;
  const int kb   = (lane >= 16) ? 16 : 0;
  const int kbB  = (lane >= 16) ? 16 : 0;

  const v8f vzero = {0.f,0.f,0.f,0.f,0.f,0.f,0.f,0.f};
  v8f acc[2][4];
#pragma unroll
  for (int i = 0; i < 2; ++i)
#pragma unroll
    for (int j = 0; j < 4; ++j) acc[i][j] = vzero;

#if USE_TDM
  const unsigned ldsA[2] = { lds_offset(&As[0][0]), lds_offset(&As[1][0]) };
  const unsigned ldsB[2] = { lds_offset(&Bs[0][0]), lds_offset(&Bs[1][0]) };
  if (wave == 0) {
    tdm_load_tile_2d(ldsA[0], A  + (size_t)gm0 * DD, 32, 128, DD);
    tdm_load_tile_2d(ldsB[0], BT + (size_t)gn0 * DD, 32, 128, DD);
  }
  for (int it = 0; it < DD / 32; ++it) {
    __builtin_amdgcn_s_wait_tensorcnt(0);   // current tile landed in LDS
    __syncthreads();                        // everyone past previous compute
    if (wave == 0 && it + 1 < DD / 32) {    // stream next tile during compute
      const int kn = (it + 1) * 32;
      const int nb = (it + 1) & 1;
      tdm_load_tile_2d(ldsA[nb], A  + (size_t)gm0 * DD + kn, 32, 128, DD);
      tdm_load_tile_2d(ldsB[nb], BT + (size_t)gn0 * DD + kn, 32, 128, DD);
    }
    const __bf16* as = As[it & 1];
    const __bf16* bs = Bs[it & 1];
#else
  for (int k0 = 0; k0 < DD; k0 += 32) {
    __syncthreads();
#pragma unroll
    for (int i = 0; i < 2; ++i) {
      int idx = i * 256 + tid;
      int row = idx >> 2;
      int q   = idx & 3;
      const __bf16* ga = A  + (size_t)(gm0 + row) * DD + k0 + q * 8;
      const __bf16* gb = BT + (size_t)(gn0 + row) * DD + k0 + q * 8;
      *(uint4*)(&As[0][row * 40 + q * 8]) = *(const uint4*)ga;
      *(uint4*)(&Bs[0][row * 40 + q * 8]) = *(const uint4*)gb;
    }
    __syncthreads();
    const __bf16* as = As[0];
    const __bf16* bs = Bs[0];
#endif

    v16bf afr[2];
#pragma unroll
    for (int mt = 0; mt < 2; ++mt)
      afr[mt] = load_afrag(&as[(wm * 32 + mt * 16 + l15) * 40], kb);
    v16bf bfr[4];
#pragma unroll
    for (int nt = 0; nt < 4; ++nt)
      bfr[nt] = load_bfrag(&bs[(wn * 64 + nt * 16 + l15) * 40], kbB);

#pragma unroll
    for (int mt = 0; mt < 2; ++mt)
#pragma unroll
      for (int nt = 0; nt < 4; ++nt)
        acc[mt][nt] = __builtin_amdgcn_wmma_f32_16x16x32_bf16(
            false, afr[mt], false, bfr[nt], (short)0, acc[mt][nt], false, false);
  }

  const int rbase = (lane >= 16) ? 8 : 0;
#pragma unroll
  for (int mt = 0; mt < 2; ++mt) {
#pragma unroll
    for (int nt = 0; nt < 4; ++nt) {
      const int n  = gn0 + wn * 64 + nt * 16 + l15;
      const float bv = bias[n];
#pragma unroll
      for (int r = 0; r < 8; ++r) {
        const int m = gm0 + wm * 32 + mt * 16 + rbase + r;
        Z[(size_t)m * GG + n] = acc[mt][nt][r] + bv;
      }
    }
  }
}

// ---------------------------------------------------------------------------
// 4) One LSTM time-step, fully fused: z = zx[:,t,:] + h @ rec, gates, c/h.
//    WG = 128 threads (4 waves).  WG tile: 64 batch rows x (16 cols x 4 gates).
//    grid = (32 j-tiles, 4 m-tiles).  TDM double-buffered: per k-step wave 0
//    issues 5 descriptors (h tile + 4 gate-block tiles of recT) for the next
//    buffer while all waves run the 4-WMMA chain on the current one.
// ---------------------------------------------------------------------------
__global__ __launch_bounds__(128) void k_lstm_step(const __bf16* __restrict__ hb_in,
                                                   const __bf16* __restrict__ RT,
                                                   const float* __restrict__ zx,
                                                   float* __restrict__ cbuf,
                                                   float* __restrict__ hbuf,
                                                   __bf16* __restrict__ hb_out,
                                                   int t) {
#if USE_TDM
  __shared__ __align__(16) __bf16 As[2][64 * 40];
  __shared__ __align__(16) __bf16 Bs[2][64 * 40];
#else
  __shared__ __align__(16) __bf16 As[1][64 * 40];
  __shared__ __align__(16) __bf16 Bs[1][64 * 40];
#endif

  const int tid  = threadIdx.x;
  const int lane = tid & 31;
  const int wave = tid >> 5;         // 0..3 -> 16-row strip
  const int m0   = blockIdx.y * 64;
  const int j0   = blockIdx.x * 16;
  const int l15  = lane & 15;
  const int kb   = (lane >= 16) ? 16 : 0;
  const int kbB  = (lane >= 16) ? 16 : 0;

  const v8f vzero = {0.f,0.f,0.f,0.f,0.f,0.f,0.f,0.f};
  v8f acc[4];
#pragma unroll
  for (int g = 0; g < 4; ++g) acc[g] = vzero;

#if USE_TDM
  const unsigned ldsA[2] = { lds_offset(&As[0][0]), lds_offset(&As[1][0]) };
  const unsigned ldsB[2] = { lds_offset(&Bs[0][0]), lds_offset(&Bs[1][0]) };
  if (wave == 0) {
    tdm_load_tile_2d(ldsA[0], hb_in + (size_t)m0 * HH, 32, 64, HH);
#pragma unroll
    for (int g = 0; g < 4; ++g)   // 16 rows of gate block g: recT[g*512 + j0 ..]
      tdm_load_tile_2d(ldsB[0] + g * 1280,
                       RT + (size_t)(g * HH + j0) * HH, 32, 16, HH);
  }
  for (int it = 0; it < HH / 32; ++it) {
    __builtin_amdgcn_s_wait_tensorcnt(0);   // current tiles landed in LDS
    __syncthreads();                        // everyone past previous compute
    if (wave == 0 && it + 1 < HH / 32) {    // stream next tiles during compute
      const int kn = (it + 1) * 32;
      const int nb = (it + 1) & 1;
      tdm_load_tile_2d(ldsA[nb], hb_in + (size_t)m0 * HH + kn, 32, 64, HH);
#pragma unroll
      for (int g = 0; g < 4; ++g)
        tdm_load_tile_2d(ldsB[nb] + g * 1280,
                         RT + (size_t)(g * HH + j0) * HH + kn, 32, 16, HH);
    }
    const __bf16* as = As[it & 1];
    const __bf16* bs = Bs[it & 1];
#else
  for (int k0 = 0; k0 < HH; k0 += 32) {
    __syncthreads();
#pragma unroll
    for (int i = 0; i < 2; ++i) {
      int idx = i * 128 + tid;
      int row = idx >> 2;
      int q   = idx & 3;
      *(uint4*)(&As[0][row * 40 + q * 8]) =
          *(const uint4*)(hb_in + (size_t)(m0 + row) * HH + k0 + q * 8);
      int nrow = (row >> 4) * HH + j0 + (row & 15);   // gate*512 + j
      *(uint4*)(&Bs[0][row * 40 + q * 8]) =
          *(const uint4*)(RT + (size_t)nrow * HH + k0 + q * 8);
    }
    __syncthreads();
    const __bf16* as = As[0];
    const __bf16* bs = Bs[0];
#endif

    v16bf afr = load_afrag(&as[(wave * 16 + l15) * 40], kb);
#pragma unroll
    for (int g = 0; g < 4; ++g) {
      v16bf bfr = load_bfrag(&bs[(g * 16 + l15) * 40], kbB);
      acc[g] = __builtin_amdgcn_wmma_f32_16x16x32_bf16(
          false, afr, false, bfr, (short)0, acc[g], false, false);
    }
  }

  const int rbase = (lane >= 16) ? 8 : 0;
  const int j = j0 + l15;
#pragma unroll
  for (int r = 0; r < 8; ++r) {
    const int b = m0 + wave * 16 + rbase + r;
    const size_t zrow = ((size_t)b * TT + t) * GG;
    const float zi = zx[zrow +          j] + acc[0][r];
    const float zf = zx[zrow +  HH    + j] + acc[1][r];
    const float zg = zx[zrow + 2 * HH + j] + acc[2][r];
    const float zo = zx[zrow + 3 * HH + j] + acc[3][r];
    const float si = 1.f / (1.f + __expf(-zi));
    const float sf = 1.f / (1.f + __expf(-zf));
    const float tg = 1.f - 2.f / (__expf(2.f * zg) + 1.f);
    const float so = 1.f / (1.f + __expf(-zo));
    const size_t ci = (size_t)b * HH + j;
    const float cn = sf * cbuf[ci] + si * tg;
    cbuf[ci] = cn;
    const float th = 1.f - 2.f / (__expf(2.f * cn) + 1.f);
    const float hv = so * th;
    hbuf[ci]   = hv;
    hb_out[ci] = (__bf16)hv;
  }
}

// ---------------------------------------------------------------------------
// 5) Head: y = leaky_relu(h@w1 + b1); out = softmax(y@w2 + b2).  1 block / b.
// ---------------------------------------------------------------------------
__global__ __launch_bounds__(128) void k_head(const float* __restrict__ h,
                                              const float* __restrict__ w1,
                                              const float* __restrict__ b1,
                                              const float* __restrict__ w2,
                                              const float* __restrict__ b2,
                                              float* __restrict__ out) {
  __shared__ float hs[HH];
  __shared__ float red[128][2];
  const int b = blockIdx.x, tid = threadIdx.x;
  for (int i = tid; i < HH; i += 128) hs[i] = h[(size_t)b * HH + i];
  __syncthreads();
  float acc = b1[tid];
  for (int k = 0; k < HH; ++k) acc = fmaf(hs[k], w1[(size_t)k * 128 + tid], acc);
  const float y = (acc > 0.f) ? acc : 0.2f * acc;
  red[tid][0] = y * w2[tid * 2 + 0];
  red[tid][1] = y * w2[tid * 2 + 1];
  __syncthreads();
  for (int s = 64; s > 0; s >>= 1) {
    if (tid < s) { red[tid][0] += red[tid + s][0]; red[tid][1] += red[tid + s][1]; }
    __syncthreads();
  }
  if (tid == 0) {
    const float l0 = red[0][0] + b2[0], l1 = red[0][1] + b2[1];
    const float m = fmaxf(l0, l1);
    const float e0 = __expf(l0 - m), e1 = __expf(l1 - m);
    const float inv = 1.f / (e0 + e1);
    out[b * 2 + 0] = e0 * inv;
    out[b * 2 + 1] = e1 * inv;
  }
}

// ---------------------------------------------------------------------------
extern "C" void kernel_launch(void* const* d_in, const int* in_sizes, int n_in,
                              void* d_out, int out_size, void* d_ws, size_t ws_size,
                              hipStream_t stream) {
  const float* x    = (const float*)d_in[0];
  const float* W    = (const float*)d_in[1];
  const float* Rw   = (const float*)d_in[2];
  const float* bias = (const float*)d_in[3];
  const float* w1   = (const float*)d_in[4];
  const float* b1   = (const float*)d_in[5];
  const float* w2   = (const float*)d_in[6];
  const float* b2   = (const float*)d_in[7];
  float* out = (float*)d_out;

  char* ws = (char*)d_ws;
  size_t off = 0;
  __bf16* xb  = (__bf16*)(ws + off); off += (size_t)MR * DD * 2;   // 52.4 MB
  __bf16* BT  = (__bf16*)(ws + off); off += (size_t)GG * DD * 2;   // 16.8 MB
  __bf16* RT  = (__bf16*)(ws + off); off += (size_t)GG * HH * 2;   //  2.1 MB
  float*  zxb = (float*) (ws + off); off += (size_t)MR * GG * 4;   // 52.4 MB
  float*  cb  = (float*) (ws + off); off += (size_t)BB * HH * 4;
  float*  hb  = (float*) (ws + off); off += (size_t)BB * HH * 4;
  __bf16* hp0 = (__bf16*)(ws + off); off += (size_t)BB * HH * 2;
  __bf16* hp1 = (__bf16*)(ws + off); off += (size_t)BB * HH * 2;

  // conversions / transposes
  k_convert_x<<<(MR * (size_t)DD) / (256 * 4), 256, 0, stream>>>(x, xb);
  k_transpose_bf16<<<dim3(GG / 32, DD / 32), 256, 0, stream>>>(W, BT, DD, GG);
  k_transpose_bf16<<<dim3(GG / 32, HH / 32), 256, 0, stream>>>(Rw, RT, HH, GG);

  // zero initial state (h0, c0)
  hipMemsetAsync(cb,  0, (size_t)BB * HH * 4, stream);
  hipMemsetAsync(hp0, 0, (size_t)BB * HH * 2, stream);

  // big input GEMM (bias fused)
  k_gemm_zx<<<dim3(GG / 128, MR / 128), 256, 0, stream>>>(xb, BT, bias, zxb);

  // 25 fused recurrence steps (ping-pong bf16 h)
  for (int t = 0; t < TT; ++t) {
    const __bf16* hin = (t & 1) ? hp1 : hp0;
    __bf16* hout      = (t & 1) ? hp0 : hp1;
    k_lstm_step<<<dim3(HH / 16, BB / 64), 128, 0, stream>>>(hin, RT, zxb, cb, hb, hout, t);
  }

  // MLP head + softmax
  k_head<<<BB, 128, 0, stream>>>(hb, w1, b1, w2, b2, out);
}
